// DWAttEncoder_8272107012821
// MI455X (gfx1250) — matrix operand order repaired and checked by reading, not compiled, not run on hardware
//
#include <hip/hip_runtime.h>

// ---------------------------------------------------------------------------
// Problem constants (from reference): L=33, D=2048, DB=1024, P=24, B=2048
// ---------------------------------------------------------------------------
#define CL  33
#define CD  2048
#define CDB 1024
#define CP  24
#define CB  2048
#define CEPS 1e-5f

typedef __bf16 bf16_t;
typedef __attribute__((ext_vector_type(16))) __bf16 v16bf;
typedef __attribute__((ext_vector_type(8)))  __bf16 v8bf;
typedef __attribute__((ext_vector_type(4)))  __bf16 v4bf;
typedef __attribute__((ext_vector_type(8)))  float  v8f;
typedef __attribute__((ext_vector_type(4)))  float  v4f;

// ---------------------------------------------------------------------------
// Tiled bf16 WMMA GEMM:  Out[z] = act(A[z] @ W[z] + bias[z])
//   A: M x K (f32 or bf16), W: K x N (f32, row-major, ld = N), Out: f32
//   Tile: BM=128 x BN=128 x BK=32, 256 threads = 8 waves, wave tile 32x64.
//   Register-prefetch pipeline: next K-tile is loaded to VGPRs while the
//   current tile's WMMAs run; f32->bf16 conversion happens on the LDS store.
// ---------------------------------------------------------------------------
#define BM 128
#define BN 128
#define BK 32
#define APAD 40   // LDS row stride (bf16 units), multiple of 8 for 16B ops
#define BPAD 40

template<bool A_BF16, bool GELU>
__global__ __launch_bounds__(256)
void gemm_bf16_wmma(const void* __restrict__ Aglob, long lda, long a_z,
                    const float* __restrict__ Wglob, long w_z,
                    const float* __restrict__ biasg, long bias_z,
                    float* __restrict__ Outg, long ldo, long out_z,
                    int M, int N, int K)
{
    __shared__ __align__(16) bf16_t Asm[BM * APAD];
    __shared__ __align__(16) bf16_t Bsm[BN * BPAD];

    const int tid  = threadIdx.x;
    const int lane = tid & 31;
    const int wid  = tid >> 5;
    const int wm   = wid & 3;     // wave row group (4 x 32 rows)
    const int wn   = wid >> 2;    // wave col group (2 x 64 cols)
    const int z    = blockIdx.z;
    const int m0   = blockIdx.y * BM;
    const int n0   = blockIdx.x * BN;

    const float* W  = Wglob + (long)z * w_z;
    const float* bv = biasg + (long)z * bias_z;
    float*       out = Outg + (long)z * out_z;
    const long   abase = (long)z * a_z;

    const int lm = lane & 15;   // row (A) / col (B) within 16
    const int lh = lane >> 4;   // K-half selector

    // Staging assignments:
    //  A: 128x32 = 4096 elems; thread owns two 8-wide row chunks.
    //  B: 32x128 = 4096 elems; thread owns one 4x4 block (micro-transpose).
    const int ar0 = ((0*256 + tid) >> 2);          // A row, chunk 0
    const int ar1 = ((1*256 + tid) >> 2);          // A row, chunk 1
    const int akc = (tid & 3) * 8;                 // A col base (8 wide)
    const int bkb = (tid >> 5) * 4;                // B k base  (4 rows)
    const int bnb = (tid & 31) * 4;                // B n base  (4 cols)

    v8f c[2][4];
    #pragma unroll
    for (int i = 0; i < 2; ++i)
        #pragma unroll
        for (int j = 0; j < 4; ++j)
            c[i][j] = (v8f){0.f,0.f,0.f,0.f,0.f,0.f,0.f,0.f};

    // ---- register prefetch buffers ----
    v4f  aRf[2][2];   // f32 A path: [chunk][4-wide half]
    v8bf aRb[2];      // bf16 A path
    v4f  bR[4];       // B rows bkb..bkb+3, 4 floats each at bnb

    auto load_tile = [&](int k0) {
        if (A_BF16) {
            const bf16_t* Ab = (const bf16_t*)Aglob + abase;
            aRb[0] = *(const v8bf*)&Ab[(long)(m0 + ar0)*lda + k0 + akc];
            aRb[1] = *(const v8bf*)&Ab[(long)(m0 + ar1)*lda + k0 + akc];
        } else {
            const float* Af = (const float*)Aglob + abase;
            #pragma unroll
            for (int h = 0; h < 2; ++h) {
                aRf[0][h] = *(const v4f*)&Af[(long)(m0 + ar0)*lda + k0 + akc + 4*h];
                aRf[1][h] = *(const v4f*)&Af[(long)(m0 + ar1)*lda + k0 + akc + 4*h];
            }
        }
        #pragma unroll
        for (int i = 0; i < 4; ++i)
            bR[i] = *(const v4f*)&W[(long)(k0 + bkb + i)*N + n0 + bnb];
        // stream hint for the W tile two steps ahead (global_prefetch_b8)
        if (k0 + 2*BK < K)
            __builtin_prefetch(&W[(long)(k0 + 2*BK + bkb)*N + n0 + bnb], 0, 1);
    };

    load_tile(0);

    for (int k0 = 0; k0 < K; k0 += BK) {
        __syncthreads();   // previous iteration's fragment reads complete
        // ---- commit prefetched tile to LDS (packed bf16 stores) ----
        if (A_BF16) {
            *(v8bf*)&Asm[ar0*APAD + akc] = aRb[0];
            *(v8bf*)&Asm[ar1*APAD + akc] = aRb[1];
        } else {
            #pragma unroll
            for (int ii = 0; ii < 2; ++ii) {
                v8bf t;
                #pragma unroll
                for (int e = 0; e < 4; ++e) {
                    t[e]   = (bf16_t)aRf[ii][0][e];
                    t[4+e] = (bf16_t)aRf[ii][1][e];
                }
                *(v8bf*)&Asm[(ii ? ar1 : ar0)*APAD + akc] = t;
            }
        }
        #pragma unroll
        for (int j = 0; j < 4; ++j) {   // 4x4 micro-transpose: Bsm[n][k]
            v4bf t;
            #pragma unroll
            for (int i = 0; i < 4; ++i) t[i] = (bf16_t)bR[i][j];
            *(v4bf*)&Bsm[(bnb + j)*BPAD + bkb] = t;
        }
        __syncthreads();

        // ---- issue next tile's global loads (overlap with WMMA) ----
        if (k0 + BK < K) load_tile(k0 + BK);

        // ---- fragments per ISA 16-bit WMMA VGPR layouts ----
        v16bf afrag[2], bfrag[4];
        #pragma unroll
        for (int ms = 0; ms < 2; ++ms) {
            // A 16x32: lane m holds K = {8h..8h+7} and {16+8h..16+8h+7}
            const bf16_t* p = &Asm[(wm*32 + ms*16 + lm)*APAD + 8*lh];
            v8bf lo = *(const v8bf*)p;
            v8bf hi = *(const v8bf*)(p + 16);
            #pragma unroll
            for (int e = 0; e < 8; ++e) { afrag[ms][e] = lo[e]; afrag[ms][8+e] = hi[e]; }
        }
        #pragma unroll
        for (int ns = 0; ns < 4; ++ns) {
            // B 32x16: lane n holds column n, K = {8h..8h+7} and {16+8h..16+8h+7}
            const bf16_t* p = &Bsm[(wn*64 + ns*16 + lm)*BPAD + 8*lh];
            v8bf lo = *(const v8bf*)p;
            v8bf hi = *(const v8bf*)(p + 16);
            #pragma unroll
            for (int e = 0; e < 8; ++e) { bfrag[ns][e] = lo[e]; bfrag[ns][8+e] = hi[e]; }
        }

        #pragma unroll
        for (int ms = 0; ms < 2; ++ms)
            #pragma unroll
            for (int ns = 0; ns < 4; ++ns)
                c[ms][ns] = __builtin_amdgcn_wmma_f32_16x16x32_bf16(
                    false, afrag[ms], false, bfrag[ns],
                    (short)0, c[ms][ns], false, false);
    }

    // ---- epilogue: bias (+ exact GELU), scatter to f32 output ----
    #pragma unroll
    for (int ms = 0; ms < 2; ++ms) {
        #pragma unroll
        for (int ns = 0; ns < 4; ++ns) {
            int col = n0 + wn*64 + ns*16 + lm;
            float bcol = bv[col];
            #pragma unroll
            for (int v = 0; v < 8; ++v) {
                int row = m0 + wm*32 + ms*16 + v + 8*lh;  // C/D layout: lanes16-31 -> M+8
                float val = c[ms][ns][v] + bcol;
                if (GELU) val = 0.5f * val * (1.0f + erff(val * 0.70710678118f));
                out[(long)row*ldo + col] = val;
            }
        }
    }
}

// ---------------------------------------------------------------------------
// Row LayerNorm (f32 in -> bf16 out), one workgroup per row.
// gamma/beta row selected by (row % lmod).
// ---------------------------------------------------------------------------
__global__ __launch_bounds__(256)
void ln_to_bf16(const float* __restrict__ in, bf16_t* __restrict__ outp,
                const float* __restrict__ gamma, const float* __restrict__ beta,
                int ncols, int lmod)
{
    __shared__ float s_sum[256];
    __shared__ float s_sq[256];
    const long r = blockIdx.x;
    const int  t = threadIdx.x;
    const int  l = (int)(r % (long)lmod);
    const float* row = in + r*(long)ncols;
    const float* g   = gamma + (long)l*ncols;
    const float* bb  = beta  + (long)l*ncols;

    float s = 0.f, sq = 0.f;
    for (int cc = t; cc < ncols; cc += 256) { float v = row[cc]; s += v; sq += v*v; }
    s_sum[t] = s; s_sq[t] = sq;
    __syncthreads();
    for (int st = 128; st > 0; st >>= 1) {
        if (t < st) { s_sum[t] += s_sum[t+st]; s_sq[t] += s_sq[t+st]; }
        __syncthreads();
    }
    float mean = s_sum[0] / (float)ncols;
    float var  = s_sq[0] / (float)ncols - mean*mean;
    float rs   = rsqrtf(var + CEPS);
    for (int cc = t; cc < ncols; cc += 256) {
        float v = (row[cc] - mean) * rs * g[cc] + bb[cc];
        outp[r*(long)ncols + cc] = (bf16_t)v;
    }
}

// ---------------------------------------------------------------------------
// keys = pos_emb (L x P) @ Wk (P x D)
// ---------------------------------------------------------------------------
__global__ __launch_bounds__(256)
void keys_kernel(const float* __restrict__ pos_emb, const float* __restrict__ Wk,
                 float* __restrict__ keysm)
{
    int idx = blockIdx.x*256 + threadIdx.x;
    if (idx >= CL*CD) return;
    int l = idx / CD, d = idx % CD;
    float acc = 0.f;
    #pragma unroll
    for (int p = 0; p < CP; ++p) acc += pos_emb[l*CP + p] * Wk[p*CD + d];
    keysm[idx] = acc;
}

// ---------------------------------------------------------------------------
// query = 1 + elu(z_L + q_tr)   (in-place on q_tr buffer)
// ---------------------------------------------------------------------------
__global__ __launch_bounds__(256)
void query_kernel(const float* __restrict__ x, float* __restrict__ q)
{
    long idx = (long)blockIdx.x*256 + threadIdx.x;
    long b = idx / CD, d = idx % CD;
    float z = x[(b*CL + (CL-1))*(long)CD + d];
    float v = z + q[idx];
    q[idx] = (v > 0.f) ? (1.f + v) : __expf(v);   // 1 + elu
}

// ---------------------------------------------------------------------------
// Final: logits = query @ keys^T, softmax over L, LN2-on-the-fly weighted sum.
// One workgroup (256 threads) per batch row; each thread owns 8 of 2048 cols.
// ---------------------------------------------------------------------------
__global__ __launch_bounds__(256)
void final_kernel(const float* __restrict__ x, const float* __restrict__ query,
                  const float* __restrict__ keysm, const float* __restrict__ vpre,
                  const float* __restrict__ g2, const float* __restrict__ b2ln,
                  float* __restrict__ outp)
{
    __shared__ float s_a[256];
    __shared__ float s_b[256];
    __shared__ float s_w[CL];
    const int b = blockIdx.x;
    const int t = threadIdx.x;

    float q[8];
    #pragma unroll
    for (int i = 0; i < 8; ++i) q[i] = query[(long)b*CD + t + i*256];

    // logits
    for (int l = 0; l < CL; ++l) {
        float p = 0.f;
        #pragma unroll
        for (int i = 0; i < 8; ++i) p += q[i] * keysm[l*CD + t + i*256];
        s_a[t] = p;
        __syncthreads();
        for (int st = 128; st > 0; st >>= 1) {
            if (t < st) s_a[t] += s_a[t+st];
            __syncthreads();
        }
        if (t == 0) s_w[l] = s_a[0];
        __syncthreads();
    }
    // softmax over 33
    if (t == 0) {
        float m = s_w[0];
        for (int l = 1; l < CL; ++l) m = fmaxf(m, s_w[l]);
        float sum = 0.f;
        for (int l = 0; l < CL; ++l) { float e = __expf(s_w[l]-m); s_w[l] = e; sum += e; }
        float inv = 1.f/sum;
        for (int l = 0; l < CL; ++l) s_w[l] *= inv;
    }
    __syncthreads();

    float acc[8];
    #pragma unroll
    for (int i = 0; i < 8; ++i) acc[i] = 0.f;

    for (int l = 0; l < CL; ++l) {
        const float* row = vpre + ((long)b*CL + l)*(long)CD;
        float v[8], s = 0.f, sq = 0.f;
        #pragma unroll
        for (int i = 0; i < 8; ++i) { v[i] = row[t + i*256]; s += v[i]; sq += v[i]*v[i]; }
        s_a[t] = s; s_b[t] = sq;
        __syncthreads();
        for (int st = 128; st > 0; st >>= 1) {
            if (t < st) { s_a[t] += s_a[t+st]; s_b[t] += s_b[t+st]; }
            __syncthreads();
        }
        float mean = s_a[0] * (1.f/(float)CD);
        float var  = s_b[0] * (1.f/(float)CD) - mean*mean;
        float rs   = rsqrtf(var + CEPS);
        float w    = s_w[l];
        #pragma unroll
        for (int i = 0; i < 8; ++i) {
            int d = t + i*256;
            acc[i] += w * ((v[i]-mean)*rs*g2[l*CD + d] + b2ln[l*CD + d]);
        }
        __syncthreads();  // protect s_a/s_b reuse next iteration
    }

    #pragma unroll
    for (int i = 0; i < 8; ++i) {
        int d = t + i*256;
        outp[(long)b*CD + d] = x[((long)b*CL + (CL-1))*(long)CD + d] + acc[i];
    }
}

// ---------------------------------------------------------------------------
extern "C" void kernel_launch(void* const* d_in, const int* in_sizes, int n_in,
                              void* d_out, int out_size, void* d_ws, size_t ws_size,
                              hipStream_t stream)
{
    const float* x      = (const float*)d_in[0];
    const float* pos    = (const float*)d_in[1];
    const float* Wk     = (const float*)d_in[2];
    const float* W1     = (const float*)d_in[3];
    const float* b1     = (const float*)d_in[4];
    const float* ln1_g  = (const float*)d_in[5];
    const float* ln1_b  = (const float*)d_in[6];
    const float* W2     = (const float*)d_in[7];
    const float* b2     = (const float*)d_in[8];
    const float* ln2_g  = (const float*)d_in[9];
    const float* ln2_b  = (const float*)d_in[10];
    const float* Wq1    = (const float*)d_in[11];
    const float* bq1    = (const float*)d_in[12];
    const float* lnq_g  = (const float*)d_in[13];
    const float* lnq_b  = (const float*)d_in[14];
    const float* Wq2    = (const float*)d_in[15];
    const float* bq2    = (const float*)d_in[16];
    float* out = (float*)d_out;
    char*  ws  = (char*)d_ws;

    // Workspace layout (with safe lifetime-based overlaps):
    //   [0, B*L*D*4)          : H (B,L,DB f32) during stage1a, then V (B,L,D f32)
    //   then Hn  (B,L,DB bf16)
    //   then Hqn (B,DB bf16)
    //   then qtr/query/Hq region (B*D f32; Hq uses first B*DB f32, dead early)
    //   then keys (L*D f32)
    size_t off = 0;
    float*  V    = (float*)(ws + off);            // also H region
    float*  H    = V;
    off += (size_t)CB*CL*CD*4;
    bf16_t* Hn   = (bf16_t*)(ws + off); off += (size_t)CB*CL*CDB*2;
    bf16_t* Hqn  = (bf16_t*)(ws + off); off += (size_t)CB*CDB*2;
    float*  qtr  = (float*)(ws + off);            // also Hq, also final query
    float*  Hq   = qtr;
    off += (size_t)CB*CD*4;
    float*  keysm = (float*)(ws + off); off += (size_t)CL*CD*4;
    (void)ws_size; (void)in_sizes; (void)n_in; (void)out_size;

    dim3 blk(256);

    // keys = pos_emb @ Wk
    keys_kernel<<<dim3((CL*CD + 255)/256), blk, 0, stream>>>(pos, Wk, keysm);

    // Stage 1a: H[:,l,:] = gelu(x[:,l,:] @ W1[l] + b1[l]),   33 slices
    gemm_bf16_wmma<false, true><<<dim3(CDB/BN, CB/BM, CL), blk, 0, stream>>>(
        x, (long)CL*CD, (long)CD,
        W1, (long)CD*CDB, b1, (long)CDB,
        H, (long)CL*CDB, (long)CDB, CB, CDB, CD);

    // Query MLP layer 1: Hq = gelu(z_L @ Wq1 + bq1)
    gemm_bf16_wmma<false, true><<<dim3(CDB/BN, CB/BM, 1), blk, 0, stream>>>(
        x + (size_t)(CL-1)*CD, (long)CL*CD, 0,
        Wq1, 0, bq1, 0,
        Hq, (long)CDB, 0, CB, CDB, CD);

    // LN1 -> bf16 (feeds next GEMM as A operand)
    ln_to_bf16<<<dim3(CB*CL), blk, 0, stream>>>(H, Hn, ln1_g, ln1_b, CDB, CL);
    ln_to_bf16<<<dim3(CB), blk, 0, stream>>>(Hq, Hqn, lnq_g, lnq_b, CDB, 1);

    // Stage 1b: V[:,l,:] = Hn[:,l,:] @ W2[l] + b2[l]  (pre-LN values; H is dead)
    gemm_bf16_wmma<true, false><<<dim3(CD/BN, CB/BM, CL), blk, 0, stream>>>(
        Hn, (long)CL*CDB, (long)CDB,
        W2, (long)CDB*CD, b2, (long)CD,
        V, (long)CL*CD, (long)CD, CB, CD, CDB);

    // Query MLP layer 2: qtr = Hqn @ Wq2 + bq2  (overwrites dead Hq region)
    gemm_bf16_wmma<true, false><<<dim3(CD/BN, CB/BM, 1), blk, 0, stream>>>(
        Hqn, (long)CDB, 0,
        Wq2, 0, bq2, 0,
        qtr, (long)CD, 0, CB, CD, CDB);

    // query = 1 + elu(z_L + qtr), in place
    query_kernel<<<dim3((size_t)CB*CD/256), blk, 0, stream>>>(x, qtr);

    // softmax attention + fused LN2 + residual
    final_kernel<<<dim3(CB), blk, 0, stream>>>(x, qtr, keysm, V, ln2_g, ln2_b, out);
}
